// ResidualMambaBlock_57045755625694
// MI455X (gfx1250) — matrix operand order repaired and checked by reading, not compiled, not run on hardware
//
#include <hip/hip_runtime.h>

#define DEV __device__ __forceinline__

typedef __attribute__((ext_vector_type(16))) __bf16 v16bf;
typedef __attribute__((ext_vector_type(8)))  __bf16 bf16x8;   // 16 bytes
typedef __attribute__((ext_vector_type(8)))  float  v8f;
typedef __attribute__((ext_vector_type(4)))  unsigned int u32x4;
typedef __attribute__((ext_vector_type(8)))  int i32x8;
typedef __attribute__((ext_vector_type(4)))  int i32x4;

#if __has_builtin(__builtin_amdgcn_tensor_load_to_lds) && __has_builtin(__builtin_amdgcn_s_wait_tensorcnt)
#define USE_TDM 1
#else
#define USE_TDM 0
#endif

// Problem sizes (fixed by the reference)
constexpr int L_  = 512;
constexpr int DM_ = 768;
constexpr int DS  = 64;
constexpr int DCV = 4;
constexpr int DI_ = 1536;
constexpr int DR_ = 48;
constexpr int N2_ = 2 * DI_;        // 3072
constexpr int XD_ = DR_ + 2 * DS;   // 176

// ---------------- helpers ----------------
DEV __bf16 f2bf(float f) {
  unsigned int u = __builtin_bit_cast(unsigned int, f);
  unsigned int r = (u + 0x7FFFu + ((u >> 16) & 1u)) >> 16;   // RNE
  return __builtin_bit_cast(__bf16, (unsigned short)r);
}
DEV float sigmoidf_(float v) { return 1.0f / (1.0f + __expf(-v)); }

#if USE_TDM
// Issue a TDM load of a [32 rows x 32 cols] bf16 tile (row stride = tensorK elems)
// from gaddr into LDS at byte offset lds_addr (packed 32x32, 64B row pitch).
DEV void tdm_load_tile(unsigned lds_addr, const __bf16* gaddr, int tensorK) {
  unsigned long long ga = (unsigned long long)(size_t)gaddr;
  u32x4 g0;
  g0[0] = 1u;                                            // count=1 (valid user D#)
  g0[1] = lds_addr;                                      // lds_addr (bytes)
  g0[2] = (unsigned)(ga & 0xFFFFFFFFull);                // global_addr[31:0]
  g0[3] = (unsigned)((ga >> 32) & 0x1FFFFFFull)          // global_addr[56:32]
        | 0x80000000u;                                   // type = 2 ("image")
  i32x8 g1;
  unsigned K = (unsigned)tensorK;
  g1[0] = (int)(1u << 16);                               // data_size=1 -> 2 bytes
  g1[1] = (int)((K & 0xFFFFu) << 16);                    // tensor_dim0[15:0]
  g1[2] = (int)((K >> 16) & 0xFFFFu)                     // tensor_dim0[31:16]
        | (int)(32u << 16);                              // tensor_dim1[15:0] = 32
  g1[3] = (int)(32u << 16);                              // tensor_dim1 hi=0, tile_dim0 = 32
  g1[4] = 32;                                            // tile_dim1 = 32, tile_dim2 = 0
  g1[5] = (int)K;                                        // tensor_dim0_stride[31:0]
  g1[6] = 0;                                             // stride hi / dim1_stride lo
  g1[7] = 0;
  i32x4 z4 = {0, 0, 0, 0};
  i32x8 z8 = {0, 0, 0, 0, 0, 0, 0, 0};
  __builtin_amdgcn_tensor_load_to_lds(g0, g1, z4, z4, z8, 0);
}
#endif

// ---------------- f32 -> bf16 cast ----------------
__global__ void k_cast_bf16(const float* __restrict__ src, __bf16* __restrict__ dst, int n) {
  int i = blockIdx.x * blockDim.x + threadIdx.x;
  if (i < n) dst[i] = f2bf(src[i]);
}

// ---------------- layernorm -> bf16 ----------------
__global__ void k_ln(const float* __restrict__ x, const float* __restrict__ w,
                     const float* __restrict__ b, __bf16* __restrict__ h_bf) {
  const int l = blockIdx.x;
  const float* xr = x + (size_t)l * DM_;
  float s = 0.f, s2 = 0.f;
  for (int i = threadIdx.x; i < DM_; i += blockDim.x) { float v = xr[i]; s += v; s2 += v * v; }
  for (int o = 16; o > 0; o >>= 1) { s += __shfl_down(s, o); s2 += __shfl_down(s2, o); }
  __shared__ float red[2][8];
  __shared__ float stat[2];
  int wv = threadIdx.x >> 5, ln = threadIdx.x & 31;
  if (ln == 0) { red[0][wv] = s; red[1][wv] = s2; }
  __syncthreads();
  if (threadIdx.x == 0) {
    float a = 0.f, a2 = 0.f;
    for (int i = 0; i < 8; i++) { a += red[0][i]; a2 += red[1][i]; }
    float mu = a / DM_;
    float var = a2 / DM_ - mu * mu;
    stat[0] = mu; stat[1] = rsqrtf(var + 1e-5f);
  }
  __syncthreads();
  float mu = stat[0], rs = stat[1];
  for (int i = threadIdx.x; i < DM_; i += blockDim.x) {
    float v = (xr[i] - mu) * rs * w[i] + b[i];
    h_bf[(size_t)l * DM_ + i] = f2bf(v);
  }
}

// ------ WMMA GEMM: C[M,N] = A[M,K](bf16) * W[N,K]^T(bf16) (+ optional add) ------
// grid.x = M/32, grid.y = ceil(N/256); block = 256 (8 waves, 2x4 grid).
// Each wave computes 16 rows x 64 cols (4 accumulators, A-fragment reuse 4x).
// A tile (32x32 bf16) is DMA'd into LDS by the Tensor Data Mover, double buffered
// with compile-time ping-pong and immediate s_wait_tensorcnt values.
template <int M, int N, int K, bool HAS_ADD>
__global__ void k_gemm_bf16(const __bf16* __restrict__ A, const __bf16* __restrict__ W,
                            const float* __restrict__ addsrc, float* __restrict__ Cout) {
  constexpr int NK = K / 32;
  static_assert(K % 64 == 0 && NK >= 2, "K layout");
  constexpr bool FULL = (N % 256) == 0;   // every 16-col tile valid in every block

  __shared__ __bf16 Atile[2][32 * 32];
  const int wave  = threadIdx.x >> 5;
  const int lane  = threadIdx.x & 31;
  const int waveM = wave >> 2;            // 0..1
  const int waveN = wave & 3;             // 0..3
  const int row0  = blockIdx.x * 32;
  const int rowT  = row0 + waveM * 16;
  const int colT  = blockIdx.y * 256 + waveN * 64;
  const int r     = lane & 15;            // row (A) / col (B) within a 16-tile
  const int hi    = lane >> 4;            // lane half-group
  const int kbA   = hi * 8;               // A-fragment interleaved K base
  const int kbB   = hi * 8;               // B-fragment interleaved K base

  v8f acc[4];
#pragma unroll
  for (int t = 0; t < 4; t++) acc[t] = (v8f){};

  auto compute = [&](const __bf16* abase, int k0) {
    // A fragment from LDS: a[i] = A[r][kbA+i] (i<8), A[r][kbA+16+(i-8)] (i>=8)
    const __bf16* arow = abase + (waveM * 16 + r) * 32;
    bf16x8 alo = *(const bf16x8*)(arow + kbA);
    bf16x8 ahi = *(const bf16x8*)(arow + kbA + 16);
    v16bf af;
#pragma unroll
    for (int i = 0; i < 8; i++) { af[i] = alo[i]; af[i + 8] = ahi[i]; }
#pragma unroll
    for (int t = 0; t < 4; t++) {
      const int c0 = colT + t * 16;       // wave-uniform
      if (FULL || c0 < N) {
        const __bf16* wrow = &W[(size_t)(c0 + r) * K + k0 + kbB];
        bf16x8 blo = *(const bf16x8*)wrow;
        bf16x8 bhi = *(const bf16x8*)(wrow + 16);
        if (k0 + 32 < K) __builtin_prefetch(wrow + 32, 0, 1);
        v16bf bfr;
#pragma unroll
        for (int i = 0; i < 8; i++) { bfr[i] = blo[i]; bfr[i + 8] = bhi[i]; }
        acc[t] = __builtin_amdgcn_wmma_f32_16x16x32_bf16(false, af, false, bfr,
                                                         (short)0, acc[t], false, false);
      }
    }
  };

#if USE_TDM
  const unsigned lds0 = (unsigned)(size_t)&Atile[0][0];
  const unsigned lds1 = (unsigned)(size_t)&Atile[1][0];
  const __bf16* Abase = &A[(size_t)row0 * K];
  if (wave == 0) tdm_load_tile(lds0, Abase, K);          // tile 0 -> buf0

  int ks = 0;
  for (; ks + 2 < NK; ks += 2) {
    if (wave == 0) {
      tdm_load_tile(lds1, Abase + (ks + 1) * 32, K);     // next -> buf1
      __builtin_amdgcn_s_wait_tensorcnt(1);              // buf0 ready
    }
    __syncthreads();
    compute(&Atile[0][0], ks * 32);
    __syncthreads();
    if (wave == 0) {
      tdm_load_tile(lds0, Abase + (ks + 2) * 32, K);     // next -> buf0
      __builtin_amdgcn_s_wait_tensorcnt(1);              // buf1 ready
    }
    __syncthreads();
    compute(&Atile[1][0], ks * 32 + 32);
    __syncthreads();
  }
  // final pair (NK even)
  if (wave == 0) {
    tdm_load_tile(lds1, Abase + (NK - 1) * 32, K);       // last -> buf1
    __builtin_amdgcn_s_wait_tensorcnt(1);                // buf0 ready
  }
  __syncthreads();
  compute(&Atile[0][0], (NK - 2) * 32);
  __syncthreads();
  if (wave == 0) __builtin_amdgcn_s_wait_tensorcnt(0);   // drain: buf1 ready
  __syncthreads();
  compute(&Atile[1][0], (NK - 1) * 32);
  __syncthreads();
#else
  for (int ks = 0; ks < NK; ++ks) {
    const int k0 = ks * 32;
    {
      int idx = threadIdx.x * 4;          // 1024 bf16 / 256 threads
      int ar = idx >> 5, ac = idx & 31;
      *(unsigned long long*)&Atile[0][idx] =
          *(const unsigned long long*)&A[(size_t)(row0 + ar) * K + k0 + ac];
    }
    __syncthreads();
    compute(&Atile[0][0], k0);
    __syncthreads();
  }
#endif

#pragma unroll
  for (int t = 0; t < 4; t++) {
    const int c0 = colT + t * 16;
    if (FULL || c0 < N) {
#pragma unroll
      for (int j = 0; j < 8; j++) {
        int m = rowT + j + hi * 8;
        int n = c0 + r;
        float v = acc[t][j];
        if (HAS_ADD) v += addsrc[(size_t)m * N + n];
        Cout[(size_t)m * N + n] = v;
      }
    }
  }
}

// ---------------- causal depthwise conv + silu ----------------
__global__ void k_conv_silu(const float* __restrict__ xz, const float* __restrict__ cw,
                            const float* __restrict__ cb, float* __restrict__ xc,
                            __bf16* __restrict__ xc_bf) {
  int t = blockIdx.x * blockDim.x + threadIdx.x;
  if (t >= L_ * DI_) return;
  int l = t / DI_, d = t % DI_;
  float acc = cb[d];
#pragma unroll
  for (int j = 0; j < DCV; j++) {
    int l2 = l - (DCV - 1) + j;
    if (l2 >= 0) acc += xz[(size_t)l2 * N2_ + d] * cw[d * DCV + j];
  }
  float y = acc * sigmoidf_(acc);     // silu
  xc[t] = y;
  xc_bf[t] = f2bf(y);
}

// ---------------- dt projection + softplus ----------------
__global__ void k_dt(const float* __restrict__ x_dbl, const float* __restrict__ dtw,
                     const float* __restrict__ dtb, float* __restrict__ dt) {
  int t = blockIdx.x * blockDim.x + threadIdx.x;
  if (t >= L_ * DI_) return;
  int l = t / DI_, d = t % DI_;
  const float* xr = x_dbl + (size_t)l * XD_;
  const float* wr = dtw + (size_t)d * DR_;
  float a = dtb[d];
#pragma unroll 8
  for (int r = 0; r < DR_; r++) a += xr[r] * wr[r];
  dt[t] = (a > 20.f) ? a : log1pf(__expf(a));
}

// ---------------- selective scan: wave per channel, 2 states per lane ----------------
__global__ void k_scan(const float* __restrict__ dt, const float* __restrict__ x_dbl,
                       const float* __restrict__ xc, const float* __restrict__ A_log,
                       float* __restrict__ y) {
  const int wave = threadIdx.x >> 5;
  const int lane = threadIdx.x & 31;
  const int d = blockIdx.x * 8 + wave;
  const int n0 = lane * 2, n1 = n0 + 1;
  const float A0 = -__expf(A_log[(size_t)d * DS + n0]);
  const float A1 = -__expf(A_log[(size_t)d * DS + n1]);
  float s0 = 0.f, s1 = 0.f;
  for (int l = 0; l < L_; l++) {
    const float* row = x_dbl + (size_t)l * XD_ + DR_;   // B at [48,112), C at [112,176)
    float Bn0 = row[n0], Bn1 = row[n1];
    float Cn0 = row[DS + n0], Cn1 = row[DS + n1];
    float dtv = dt[(size_t)l * DI_ + d];
    float xv  = xc[(size_t)l * DI_ + d];
    float dbx = dtv * xv;
    s0 = __expf(dtv * A0) * s0 + dbx * Bn0;
    s1 = __expf(dtv * A1) * s1 + dbx * Bn1;
    float part = s0 * Cn0 + s1 * Cn1;
    for (int o = 16; o > 0; o >>= 1) part += __shfl_down(part, o);
    if (lane == 0) y[(size_t)l * DI_ + d] = part;
  }
}

// ---------------- y = (y + xc*D) * silu(z) -> bf16 ----------------
__global__ void k_gate(const float* __restrict__ y, const float* __restrict__ xc,
                       const float* __restrict__ Dp, const float* __restrict__ xz,
                       __bf16* __restrict__ ybf) {
  int t = blockIdx.x * blockDim.x + threadIdx.x;
  if (t >= L_ * DI_) return;
  int l = t / DI_, d = t % DI_;
  float z = xz[(size_t)l * N2_ + DI_ + d];
  float v = (y[t] + xc[t] * Dp[d]) * (z * sigmoidf_(z));
  ybf[t] = f2bf(v);
}

// ---------------- host launcher ----------------
extern "C" void kernel_launch(void* const* d_in, const int* in_sizes, int n_in,
                              void* d_out, int out_size, void* d_ws, size_t ws_size,
                              hipStream_t stream) {
  const float* x         = (const float*)d_in[0];
  const float* ln_w      = (const float*)d_in[1];
  const float* ln_b      = (const float*)d_in[2];
  const float* in_proj_w = (const float*)d_in[3];
  const float* conv_w    = (const float*)d_in[4];
  const float* conv_b    = (const float*)d_in[5];
  const float* x_proj_w  = (const float*)d_in[6];
  const float* dt_proj_w = (const float*)d_in[7];
  const float* dt_proj_b = (const float*)d_in[8];
  const float* A_log     = (const float*)d_in[9];
  const float* Dp        = (const float*)d_in[10];
  const float* out_proj_w= (const float*)d_in[11];
  float* out = (float*)d_out;

  char* ws = (char*)d_ws;
  size_t off = 0;
  auto alloc = [&](size_t bytes) -> char* {
    char* p = ws + off;
    off = (off + bytes + 255) & ~(size_t)255;
    return p;
  };
  __bf16* Wbf_in  = (__bf16*)alloc(sizeof(__bf16) * (size_t)N2_ * DM_);
  __bf16* Wbf_x   = (__bf16*)alloc(sizeof(__bf16) * (size_t)XD_ * DI_);
  __bf16* Wbf_out = (__bf16*)alloc(sizeof(__bf16) * (size_t)DM_ * DI_);
  __bf16* h_bf    = (__bf16*)alloc(sizeof(__bf16) * (size_t)L_ * DM_);
  float*  xz      = (float*) alloc(sizeof(float)  * (size_t)L_ * N2_);
  float*  xc      = (float*) alloc(sizeof(float)  * (size_t)L_ * DI_);
  __bf16* xc_bf   = (__bf16*)alloc(sizeof(__bf16) * (size_t)L_ * DI_);
  float*  x_dbl   = (float*) alloc(sizeof(float)  * (size_t)L_ * XD_);
  float*  dt      = (float*) alloc(sizeof(float)  * (size_t)L_ * DI_);
  float*  yb      = (float*) alloc(sizeof(float)  * (size_t)L_ * DI_);
  __bf16* y_bf    = (__bf16*)alloc(sizeof(__bf16) * (size_t)L_ * DI_);

  const int TB = 256;
  // weight casts (deterministic each call)
  {
    int n = N2_ * DM_;
    k_cast_bf16<<<(n + TB - 1) / TB, TB, 0, stream>>>(in_proj_w, Wbf_in, n);
    n = XD_ * DI_;
    k_cast_bf16<<<(n + TB - 1) / TB, TB, 0, stream>>>(x_proj_w, Wbf_x, n);
    n = DM_ * DI_;
    k_cast_bf16<<<(n + TB - 1) / TB, TB, 0, stream>>>(out_proj_w, Wbf_out, n);
  }
  // layernorm
  k_ln<<<L_, TB, 0, stream>>>(x, ln_w, ln_b, h_bf);
  // in_proj: [512,768] x [3072,768]^T -> [512,3072]
  k_gemm_bf16<512, N2_, DM_, false><<<dim3(L_ / 32, N2_ / 256), TB, 0, stream>>>(
      h_bf, Wbf_in, nullptr, xz);
  // conv + silu
  k_conv_silu<<<(L_ * DI_ + TB - 1) / TB, TB, 0, stream>>>(xz, conv_w, conv_b, xc, xc_bf);
  // x_proj: [512,1536] x [176,1536]^T -> [512,176]
  k_gemm_bf16<512, XD_, DI_, false><<<dim3(L_ / 32, (XD_ + 255) / 256), TB, 0, stream>>>(
      xc_bf, Wbf_x, nullptr, x_dbl);
  // dt projection + softplus
  k_dt<<<(L_ * DI_ + TB - 1) / TB, TB, 0, stream>>>(x_dbl, dt_proj_w, dt_proj_b, dt);
  // selective scan
  k_scan<<<DI_ / 8, TB, 0, stream>>>(dt, x_dbl, xc, A_log, yb);
  // gate
  k_gate<<<(L_ * DI_ + TB - 1) / TB, TB, 0, stream>>>(yb, xc, Dp, xz, y_bf);
  // out_proj + residual: [512,1536] x [768,1536]^T + x -> out
  k_gemm_bf16<512, DM_, DI_, true><<<dim3(L_ / 32, DM_ / 256), TB, 0, stream>>>(
      y_bf, Wbf_out, x, out);
}
